// SE3TransformerEncoder_56006373540427
// MI455X (gfx1250) — compile-verified, matrix operand order checked by reading, not run on hardware
//
#include <hip/hip_runtime.h>
#include <math.h>

#define NNODES 4096
#define NEDGES 65536
#define INDEG  16
#define RCHUNK 16384
#define NCH    16
#define CMID   4
#define CFINAL 64

typedef _Float16 v16h __attribute__((ext_vector_type(16)));
typedef float    v8f  __attribute__((ext_vector_type(8)));

// ---------------------------------------------------------------- WMMA helpers
__device__ __forceinline__ int kmap16(int j, int lane) {
  // 16-bit A/B fragment K index for element j of a v16h, per CDNA5 ISA layout
  return (j & 7) + ((j >> 3) << 4) + (((lane >> 4) & 1) << 3);
}
__device__ __forceinline__ v8f wmma_ff16(v16h a, v16h b, v8f c) {
  return __builtin_amdgcn_wmma_f32_16x16x32_f16(false, a, false, b, (short)0, c,
                                                false, false);
}

// ---------------------------------------------------------------- CG tables
__host__ __device__ inline int cg_pair_off(int a, int b) {
  int off = 0;
  for (int aa = 0; aa < 4; ++aa)
    for (int bb = 0; bb < 4; ++bb) {
      if (aa == a && bb == b) return off;
      int lo = aa > bb ? aa - bb : bb - aa;
      int hi = aa + bb;
      off += (2 * aa + 1) * (2 * bb + 1) * ((hi + 1) * (hi + 1) - lo * lo);
    }
  return off;
}

__device__ double dfact(int n) {
  double r = 1.0;
  for (int i = 2; i <= n; ++i) r *= (double)i;
  return r;
}

__device__ double dev_su2cg(int j1, int m1, int j2, int m2, int j3, int m3) {
  if (m3 != m1 + m2) return 0.0;
  double pref = (2 * j3 + 1) * dfact(j3 + j1 - j2) * dfact(j3 - j1 + j2) *
                dfact(j1 + j2 - j3) / dfact(j1 + j2 + j3 + 1);
  pref *= dfact(j3 + m3) * dfact(j3 - m3) * dfact(j1 - m1) * dfact(j1 + m1) *
          dfact(j2 - m2) * dfact(j2 + m2);
  pref = sqrt(pref);
  int kmin = 0;
  if (j2 - j3 - m1 > kmin) kmin = j2 - j3 - m1;
  if (j1 - j3 + m2 > kmin) kmin = j1 - j3 + m2;
  int kmax = j1 + j2 - j3;
  if (j1 - m1 < kmax) kmax = j1 - m1;
  if (j2 + m2 < kmax) kmax = j2 + m2;
  double s = 0.0;
  for (int k = kmin; k <= kmax; ++k) {
    double denom = dfact(k) * dfact(j1 + j2 - j3 - k) * dfact(j1 - m1 - k) *
                   dfact(j2 + m2 - k) * dfact(j3 - j2 + m1 + k) *
                   dfact(j3 - j1 - m2 + k);
    s += ((k & 1) ? -1.0 : 1.0) / denom;
  }
  return pref * s;
}

__device__ void dev_qent(int l, int row, int col, double& re, double& im) {
  const double iv = 0.7071067811865476;
  double qr = 0.0, qi = 0.0;
  int m = row - l;
  if (m < 0) {
    if (col == l - m) qr = iv;          // l + |m|
    else if (col == l + m) qi = -iv;    // l - |m|
  } else if (m == 0) {
    if (col == l) qr = 1.0;
  } else {
    double sgn = (m & 1) ? -1.0 : 1.0;
    if (col == l + m) qr = sgn * iv;
    else if (col == l - m) qi = sgn * iv;
  }
  double pr, pi;  // (-1j)^l
  switch (l & 3) {
    case 0: pr = 1; pi = 0; break;
    case 1: pr = 0; pi = -1; break;
    case 2: pr = -1; pi = 0; break;
    default: pr = 0; pi = 1; break;
  }
  re = qr * pr - qi * pi;
  im = qr * pi + qi * pr;
}

__global__ void cg_init_kernel(float* __restrict__ cg) {
  int a = blockIdx.x >> 2, b = blockIdx.x & 3;
  int base = cg_pair_off(a, b);
  int joff = 0;
  int jlo = a > b ? a - b : b - a;
  for (int J = jlo; J <= a + b; ++J) {
    int IA = 2 * a + 1, MJ = 2 * J + 1, IB = 2 * b + 1;
    int sz = IA * MJ * IB;
    for (int idx = threadIdx.x; idx < sz; idx += blockDim.x) {
      int jj = idx / (MJ * IB);
      int rem = idx - jj * MJ * IB;
      int ll = rem / IB, nn = rem - ll * IB;
      double sr = 0.0;
      for (int i = 0; i < IA; ++i)
        for (int k = 0; k < MJ; ++k) {
          int m3 = i + k - a - J + b;
          if (m3 < 0 || m3 > 2 * b) continue;
          double cgv = dev_su2cg(a, i - a, J, k - J, b, (i - a) + (k - J));
          if (cgv == 0.0) continue;
          double q1r, q1i, q2r, q2i, q3r, q3i;
          dev_qent(a, i, jj, q1r, q1i);
          dev_qent(J, k, ll, q2r, q2i);
          dev_qent(b, m3, nn, q3r, q3i);
          double tr = q1r * q2r - q1i * q2i;
          double ti = q1r * q2i + q1i * q2r;
          sr += (tr * q3r + ti * q3i) * cgv;  // * conj(q3), real part
        }
      cg[base + joff + idx] = (float)sr;
    }
    joff += sz;
  }
}

// ---------------------------------------------------------------- utils
__global__ void zero_kernel(float* p, int n) {
  int i = blockIdx.x * blockDim.x + threadIdx.x;
  if (i < n) p[i] = 0.f;
}

// ---------------------------------------------------------------- geometry: Y + efeat
__global__ void geom_kernel(const float* __restrict__ x, const int* __restrict__ src,
                            const int* __restrict__ dst, const float* __restrict__ w,
                            float* __restrict__ Y, float* __restrict__ ef) {
  int e = blockIdx.x * blockDim.x + threadIdx.x;
  if (e >= NEDGES) return;
  int s = src[e], d = dst[e];
  float dx = x[d * 3 + 0] - x[s * 3 + 0];
  float dy = x[d * 3 + 1] - x[s * 3 + 1];
  float dz = x[d * 3 + 2] - x[s * 3 + 2];
  float r = sqrtf(dx * dx + dy * dy + dz * dz + 1e-12f);
  float ux = dx / r, uy = dy / r, uz = dz / r;
  float z = uz;
  float rho = sqrtf(fmaxf(1.f - z * z, 1e-12f));
  float cphi = ux / rho, sphi = uy / rho;
  float cosm[7], sinm[7];
  cosm[0] = 1.f; sinm[0] = 0.f; cosm[1] = cphi; sinm[1] = sphi;
#pragma unroll
  for (int m = 2; m <= 6; ++m) {
    cosm[m] = cphi * cosm[m - 1] - sphi * sinm[m - 1];
    sinm[m] = sphi * cosm[m - 1] + cphi * sinm[m - 1];
  }
  float P[7][7];
  P[0][0] = 1.f;
#pragma unroll
  for (int m = 1; m <= 6; ++m) P[m][m] = -(2.f * m - 1.f) * rho * P[m - 1][m - 1];
#pragma unroll
  for (int m = 0; m <= 5; ++m) P[m + 1][m] = (2.f * m + 1.f) * z * P[m][m];
#pragma unroll
  for (int l = 2; l <= 6; ++l)
#pragma unroll
    for (int m = 0; m <= 4; ++m)
      if (m <= l - 2)
        P[l][m] = ((2.f * l - 1.f) * z * P[l - 1][m] - (l + m - 1.f) * P[l - 2][m]) /
                  (float)(l - m);
  const float FOURPI = 12.566370614359172f;
  const float SQRT2 = 1.4142135623730951f;
#pragma unroll
  for (int l = 0; l <= 6; ++l)
#pragma unroll
    for (int mm = -6; mm <= 6; ++mm) {
      if (mm < -l || mm > l) continue;
      int a = mm < 0 ? -mm : mm;
      float fr = 1.f;
      for (int t = l - a + 1; t <= l + a; ++t) fr *= (float)t;
      float Kn = sqrtf((2.f * l + 1.f) / (FOURPI * fr));
      float v;
      if (mm == 0) v = Kn * P[l][0];
      else if (mm > 0) v = SQRT2 * Kn * cosm[mm] * P[l][mm];
      else v = SQRT2 * Kn * sinm[a] * P[l][a];
      Y[(size_t)e * 49 + l * l + (l + mm)] = v;
    }
  ef[e * 8 + 0] = w[e * 4 + 0];
  ef[e * 8 + 1] = w[e * 4 + 1];
  ef[e * 8 + 2] = w[e * 4 + 2];
  ef[e * 8 + 3] = w[e * 4 + 3];
  ef[e * 8 + 4] = r;
  ef[e * 8 + 5] = 0.f; ef[e * 8 + 6] = 0.f; ef[e * 8 + 7] = 0.f;
}

// ---------------------------------------------------------------- embed (WMMA)
__global__ void __launch_bounds__(256) embed_kernel(const float* __restrict__ f,
                                                    const float* __restrict__ W,
                                                    const float* __restrict__ bb,
                                                    float* __restrict__ h) {
  int tid = threadIdx.x;
  int wave = tid >> 5, lane = tid & 31;
  int row = lane & 15, hi8 = ((lane >> 4) & 1) << 3;
  int nodeBase = blockIdx.x * 128 + wave * 16;
  v16h a, bf;
#pragma unroll
  for (int j = 0; j < 16; ++j) {
    int K = kmap16(j, lane);
    a[j] = (_Float16)f[(nodeBase + row) * 32 + K];
    bf[j] = (_Float16)W[row * 32 + K];  // W[c][k], c = col N
  }
  v8f c = {};
  c = wmma_ff16(a, bf, c);
#pragma unroll
  for (int v = 0; v < 8; ++v) {
    int n = nodeBase + v + hi8;
    h[(size_t)n * 256 + row * 16 + 0] = c[v] + bb[row];
  }
}

// ---------------------------------------------------------------- radial MLP H2 (WMMA x4)
__global__ void __launch_bounds__(256) radial_h2_kernel(
    const float* __restrict__ ef, const float* __restrict__ W1,
    const float* __restrict__ b1, const float* __restrict__ W2,
    const float* __restrict__ b2, float* __restrict__ H2) {
  __shared__ _Float16 sW1[32 * 32];
  __shared__ _Float16 sW2[32 * 32];
  __shared__ float sb1[32], sb2[32];
  __shared__ _Float16 sH1[8][16 * 32];
  int tid = threadIdx.x;
  for (int i = tid; i < 32 * 32; i += 256) {
    int hh = i >> 5, k = i & 31;
    sW1[i] = (_Float16)(k < 5 ? W1[hh * 5 + k] : 0.f);
    sW2[i] = (_Float16)W2[i];
  }
  if (tid < 32) { sb1[tid] = b1[tid]; sb2[tid] = b2[tid]; }
  __syncthreads();
  int wave = tid >> 5, lane = tid & 31;
  int row = lane & 15, hi8 = ((lane >> 4) & 1) << 3;
  int eBase = blockIdx.x * 128 + wave * 16;
  v16h a;
#pragma unroll
  for (int j = 0; j < 16; ++j) {
    int K = kmap16(j, lane);
    a[j] = (_Float16)((K < 8) ? ef[(size_t)(eBase + row) * 8 + K] : 0.f);
  }
#pragma unroll
  for (int nt = 0; nt < 2; ++nt) {
    v16h bf;
#pragma unroll
    for (int j = 0; j < 16; ++j) bf[j] = sW1[(nt * 16 + row) * 32 + kmap16(j, lane)];
    v8f c = {};
    c = wmma_ff16(a, bf, c);
#pragma unroll
    for (int v = 0; v < 8; ++v) {
      float val = fmaxf(c[v] + sb1[nt * 16 + row], 0.f);
      sH1[wave][(v + hi8) * 32 + nt * 16 + row] = (_Float16)val;
    }
  }
  v16h a2;
#pragma unroll
  for (int j = 0; j < 16; ++j) a2[j] = sH1[wave][row * 32 + kmap16(j, lane)];
#pragma unroll
  for (int nt = 0; nt < 2; ++nt) {
    v16h bf;
#pragma unroll
    for (int j = 0; j < 16; ++j) bf[j] = sW2[(nt * 16 + row) * 32 + kmap16(j, lane)];
    v8f c = {};
    c = wmma_ff16(a2, bf, c);
#pragma unroll
    for (int v = 0; v < 8; ++v) {
      float val = fmaxf(c[v] + sb2[nt * 16 + row], 0.f);
      H2[(size_t)(eBase + v + hi8) * 32 + nt * 16 + row] = val;
    }
  }
}

// ---------------------------------------------------------------- R = H2 @ W3^T + b3 (WMMA)
__global__ void __launch_bounds__(256) radial_r_kernel(
    const float* __restrict__ H2, const float* __restrict__ W3,
    const float* __restrict__ b3, float* __restrict__ R, int cols, int e0) {
  __shared__ _Float16 sW3[448 * 32];
  __shared__ float sb3[448];
  int tid = threadIdx.x;
  for (int i = tid; i < cols * 32; i += 256) sW3[i] = (_Float16)W3[i];
  for (int i = tid; i < cols; i += 256) sb3[i] = b3[i];
  __syncthreads();
  int wave = tid >> 5, lane = tid & 31;
  int row = lane & 15, hi8 = ((lane >> 4) & 1) << 3;
  int eLoc = blockIdx.x * 128 + wave * 16;
  int eG = e0 + eLoc;
  v16h a;
#pragma unroll
  for (int j = 0; j < 16; ++j)
    a[j] = (_Float16)H2[(size_t)(eG + row) * 32 + kmap16(j, lane)];
  int nt = (cols + 15) >> 4;
  for (int t = 0; t < nt; ++t) {
    int col = t * 16 + row;
    v16h bf;
#pragma unroll
    for (int j = 0; j < 16; ++j)
      bf[j] = (col < cols) ? sW3[col * 32 + kmap16(j, lane)] : (_Float16)0.f;
    v8f c = {};
    c = wmma_ff16(a, bf, c);
    if (col < cols) {
      float bv = sb3[col];
#pragma unroll
      for (int v = 0; v < 8; ++v)
        R[(size_t)(eLoc + v + hi8) * cols + col] = c[v] + bv;
    }
  }
}

// ---------------------------------------------------------------- pair conv (VALU, templated on degrees)
template <int A, int B>
__global__ void conv_pair_kernel(const float* __restrict__ R, const float* __restrict__ Y,
                                 const float* __restrict__ h, const int* __restrict__ src,
                                 const float* __restrict__ cg, float* __restrict__ out,
                                 int e0, int nE) {
  constexpr int IA = 2 * A + 1, IB = 2 * B + 1;
  constexpr int NF = 2 * (A < B ? A : B) + 1;
  constexpr int JLO = (A > B ? A - B : B - A);
  constexpr int COLS = 64 * NF;
  int el = blockIdx.x * blockDim.x + threadIdx.x;
  if (el >= nE) return;
  int e = e0 + el;
  int s = src[e];
  float ha[16][IA];
#pragma unroll
  for (int c = 0; c < 16; ++c)
#pragma unroll
    for (int i = 0; i < IA; ++i) ha[c][i] = h[(size_t)s * 256 + c * 16 + A * A + i];
  float acc[4][IB];
#pragma unroll
  for (int u = 0; u < 4; ++u)
#pragma unroll
    for (int o = 0; o < IB; ++o) acc[u][o] = 0.f;
#pragma unroll
  for (int u = 0; u < 4; ++u) {
    float Q[IA][NF];
#pragma unroll
    for (int i = 0; i < IA; ++i)
#pragma unroll
      for (int fi = 0; fi < NF; ++fi) Q[i][fi] = 0.f;
#pragma unroll
    for (int c = 0; c < 16; ++c)
#pragma unroll
      for (int fi = 0; fi < NF; ++fi) {
        float rv = R[(size_t)el * COLS + (u * 16 + c) * NF + fi];
#pragma unroll
        for (int i = 0; i < IA; ++i) Q[i][fi] += rv * ha[c][i];
      }
    int cgo = 0;
#pragma unroll
    for (int fi = 0; fi < NF; ++fi) {
      constexpr int dummy = 0; (void)dummy;
      int J = JLO + fi;
      int MJ = 2 * J + 1;
      for (int m = 0; m < MJ; ++m) {
        float y = Y[(size_t)e * 49 + J * J + m];
#pragma unroll
        for (int i = 0; i < IA; ++i) {
          float qv = Q[i][fi] * y;
#pragma unroll
          for (int o = 0; o < IB; ++o)
            acc[u][o] += qv * cg[cgo + (i * MJ + m) * IB + o];
        }
      }
      cgo += IA * MJ * IB;
    }
  }
#pragma unroll
  for (int u = 0; u < 4; ++u)
#pragma unroll
    for (int o = 0; o < IB; ++o) {
      size_t idx = (size_t)e * 64 + u * 16 + B * B + o;
      out[idx] = out[idx] + acc[u][o];
    }
}

// ---------------------------------------------------------------- q projection
__global__ void q_kernel(const float* __restrict__ h, const float* q0, const float* q1,
                         const float* q2, const float* q3, float* __restrict__ qb,
                         int ndin) {
  int n = blockIdx.x * blockDim.x + threadIdx.x;
  if (n >= NNODES) return;
  const float* qs[4] = {q0, q1, q2, q3};
  for (int i = 0; i < 64; ++i) qb[(size_t)n * 64 + i] = 0.f;
  for (int d = 0; d < ndin; ++d) {
    int ia = 2 * d + 1;
    for (int o = 0; o < 4; ++o)
      for (int i = 0; i < ia; ++i) {
        float sum = 0.f;
        for (int c = 0; c < 16; ++c)
          sum += h[(size_t)n * 256 + c * 16 + d * d + i] * qs[d][o * 16 + c];
        qb[(size_t)n * 64 + o * 16 + d * d + i] = sum;
      }
  }
}

// ---------------------------------------------------------------- attention logits
__global__ void logit_kernel(const float* __restrict__ kb, const float* __restrict__ qb,
                             const int* __restrict__ dst, float* __restrict__ lg,
                             int mtot, float inv) {
  int e = blockIdx.x * blockDim.x + threadIdx.x;
  if (e >= NEDGES) return;
  int dn = dst[e];
  float l0 = 0.f, l1 = 0.f;
  for (int u = 0; u < 4; ++u) {
    float s = 0.f;
    for (int mf = 0; mf < mtot; ++mf)
      s += kb[(size_t)e * 64 + u * 16 + mf] * qb[(size_t)dn * 64 + u * 16 + mf];
    if (u < 2) l0 += s; else l1 += s;
  }
  lg[e * 2 + 0] = l0 * inv;
  lg[e * 2 + 1] = l1 * inv;
}

__global__ void softmax_kernel(float* __restrict__ lg) {
  int n = blockIdx.x * blockDim.x + threadIdx.x;
  if (n >= NNODES) return;
  int base = n * INDEG;
#pragma unroll
  for (int hd = 0; hd < 2; ++hd) {
    float mx = -1e30f;
    for (int k = 0; k < INDEG; ++k) mx = fmaxf(mx, lg[(base + k) * 2 + hd]);
    float sm = 0.f;
    float ex[INDEG];
    for (int k = 0; k < INDEG; ++k) { ex[k] = expf(lg[(base + k) * 2 + hd] - mx); sm += ex[k]; }
    float r = 1.f / sm;
    for (int k = 0; k < INDEG; ++k) lg[(base + k) * 2 + hd] = ex[k] * r;
  }
}

// ---------------------------------------------------------------- aggregate + proj + residual + norm
template <int D>
__global__ void agg_kernel(const float* __restrict__ vb, const float* __restrict__ alpha,
                           const float* __restrict__ hin, const float* __restrict__ proj,
                           const float* __restrict__ nW, const float* __restrict__ nb,
                           float* __restrict__ hout, int ndin) {
  int n = blockIdx.x * blockDim.x + threadIdx.x;
  if (n >= NNODES) return;
  constexpr int M = 2 * D + 1;
  float agg[4][M];
#pragma unroll
  for (int u = 0; u < 4; ++u)
#pragma unroll
    for (int m = 0; m < M; ++m) agg[u][m] = 0.f;
  for (int k = 0; k < INDEG; ++k) {
    int e = n * INDEG + k;
    float a0 = alpha[e * 2 + 0], a1 = alpha[e * 2 + 1];
#pragma unroll
    for (int u = 0; u < 4; ++u) {
      float al = (u < 2) ? a0 : a1;
#pragma unroll
      for (int m = 0; m < M; ++m)
        agg[u][m] += al * vb[(size_t)e * 64 + u * 16 + D * D + m];
    }
  }
  float hn[16][M];
#pragma unroll
  for (int c = 0; c < 16; ++c)
#pragma unroll
    for (int m = 0; m < M; ++m) {
      float s = 0.f;
#pragma unroll
      for (int u = 0; u < 4; ++u) s += agg[u][m] * proj[c * 4 + u];
      hn[c][m] = s;
    }
  if (D < ndin) {
#pragma unroll
    for (int c = 0; c < 16; ++c)
#pragma unroll
      for (int m = 0; m < M; ++m)
        hn[c][m] += hin[(size_t)n * 256 + c * 16 + D * D + m];
  }
  float nm[16];
#pragma unroll
  for (int c = 0; c < 16; ++c) {
    float s = 1e-12f;
#pragma unroll
    for (int m = 0; m < M; ++m) s += hn[c][m] * hn[c][m];
    nm[c] = sqrtf(s);
  }
#pragma unroll
  for (int c = 0; c < 16; ++c) {
    float s = nb[c];
#pragma unroll
    for (int c2 = 0; c2 < 16; ++c2) s += nW[c * 16 + c2] * nm[c2];
    float scale = fmaxf(s, 0.f) / nm[c];
#pragma unroll
    for (int m = 0; m < M; ++m)
      hout[(size_t)n * 256 + c * 16 + D * D + m] = hn[c][m] * scale;
  }
}

// ---------------------------------------------------------------- final: tmp[c] per edge
template <int A>
__global__ void tmp_kernel(const float* __restrict__ h, const int* __restrict__ src,
                           const float* __restrict__ Y, const float* __restrict__ cg,
                           float* __restrict__ tb) {
  int e = blockIdx.x * blockDim.x + threadIdx.x;
  if (e >= NEDGES) return;
  constexpr int IA = 2 * A + 1;
  int s = src[e];
  float bas[IA];
#pragma unroll
  for (int i = 0; i < IA; ++i) {
    float t = 0.f;
#pragma unroll
    for (int m = 0; m < IA; ++m)
      t += Y[(size_t)e * 49 + A * A + m] * cg[i * IA + m];
    bas[i] = t;
  }
#pragma unroll
  for (int c = 0; c < 16; ++c) {
    float t = 0.f;
#pragma unroll
    for (int i = 0; i < IA; ++i) t += h[(size_t)s * 256 + c * 16 + A * A + i] * bas[i];
    tb[(size_t)e * 16 + c] = t;
  }
}

// ---------------------------------------------------------------- final outer-product GEMM (WMMA, K=512)
__global__ void __launch_bounds__(256) final_outer_kernel(
    const float* __restrict__ H2, const float* __restrict__ tb,
    const float* __restrict__ W3, const float* __restrict__ b3,
    float* __restrict__ msg) {
  int tid = threadIdx.x;
  int wave = tid >> 5, lane = tid & 31;
  int row = lane & 15, hi8 = ((lane >> 4) & 1) << 3;
  int eBase = blockIdx.x * 128 + wave * 16;
  v8f acc[4];
#pragma unroll
  for (int nt = 0; nt < 4; ++nt)
#pragma unroll
    for (int v = 0; v < 8; ++v) acc[nt][v] = 0.f;
  int eRow = eBase + row;
  for (int kc = 0; kc < 16; ++kc) {
    v16h a;
#pragma unroll
    for (int j = 0; j < 16; ++j) {
      int kk = kc * 32 + kmap16(j, lane);
      a[j] = (_Float16)(H2[(size_t)eRow * 32 + (kk >> 4)] * tb[(size_t)eRow * 16 + (kk & 15)]);
    }
#pragma unroll
    for (int nt = 0; nt < 4; ++nt) {
      v16h bf;
#pragma unroll
      for (int j = 0; j < 16; ++j) {
        int kk = kc * 32 + kmap16(j, lane);
        int u = nt * 16 + row;
        bf[j] = (_Float16)W3[(size_t)(u * 16 + (kk & 15)) * 32 + (kk >> 4)];
      }
      acc[nt] = wmma_ff16(a, bf, acc[nt]);
    }
  }
#pragma unroll
  for (int nt = 0; nt < 4; ++nt) {
    int u = nt * 16 + row;
#pragma unroll
    for (int v = 0; v < 8; ++v) {
      int e = eBase + v + hi8;
      float bias = 0.f;
      for (int cc = 0; cc < 16; ++cc)
        bias += b3[u * 16 + cc] * tb[(size_t)e * 16 + cc];
      size_t idx = (size_t)e * 64 + u;
      msg[idx] = msg[idx] + acc[nt][v] + bias;
    }
  }
}

// ---------------------------------------------------------------- final node reduce + global mean
__global__ void final_node_kernel(const float* __restrict__ msg, const float* __restrict__ h,
                                  const float* __restrict__ selfW, float* __restrict__ out) {
  int n = blockIdx.x * blockDim.x + threadIdx.x;
  if (n >= NNODES) return;
  for (int u = 0; u < 64; ++u) {
    float s = 0.f;
    for (int k = 0; k < INDEG; ++k) s += msg[(size_t)(n * INDEG + k) * 64 + u];
    s *= (1.f / (float)INDEG);
    float se = 0.f;
    for (int c = 0; c < 16; ++c) se += h[(size_t)n * 256 + c * 16 + 0] * selfW[u * 16 + c];
    atomicAdd(&out[u], (s + se) * (1.f / (float)NNODES));
  }
}

// ================================================================ host
extern "C" void kernel_launch(void* const* d_in, const int* in_sizes, int n_in,
                              void* d_out, int out_size, void* d_ws, size_t ws_size,
                              hipStream_t stream) {
  (void)in_sizes; (void)n_in; (void)out_size; (void)ws_size;
  auto in = [&](int i) { return (const float*)d_in[i]; };
  const float* fx = in(0);
  const float* xp = in(1);
  const float* wp = in(2);
  const int* srcp = (const int*)d_in[3];
  const int* dstp = (const int*)d_in[4];
  // pytree (sorted-key) flatten indices
  const int EMBW = 5, EMBB = 6;
  auto finalPair = [&](int a) { return 7 + a * 6; };
  const int SELFW = 31;
  const int KB[2] = {32, 75};
  const int NORMB[2] = {38, 171};
  const int PROJB[2] = {46, 179};
  const int QB[2] = {50, 183};
  const int VB[2] = {51, 187};
  struct RP { const float *W1, *W2, *W3, *b1, *b2, *b3; };
  auto rp = [&](int base) {
    RP r{in(base), in(base + 1), in(base + 2), in(base + 3), in(base + 4), in(base + 5)};
    return r;
  };

  // workspace layout (floats)
  float* W = (float*)d_ws;
  size_t off = 0;
  auto alloc = [&](size_t n) { float* p = W + off; off += n; return p; };
  float* cgw  = alloc(8192);
  float* Yb   = alloc((size_t)NEDGES * 49);
  float* efb  = alloc((size_t)NEDGES * 8);
  float* hA   = alloc((size_t)NNODES * 256);
  float* hB   = alloc((size_t)NNODES * 256);
  float* H2b  = alloc((size_t)NEDGES * 32);
  float* vbuf = alloc((size_t)NEDGES * 64);
  float* kbuf = alloc((size_t)NEDGES * 64);
  float* qbuf = alloc((size_t)NNODES * 64);
  float* lgb  = alloc((size_t)NEDGES * 2);
  float* tmpb = alloc((size_t)NEDGES * 16);
  float* msgb = alloc((size_t)NEDGES * 64);
  float* Rbuf = alloc((size_t)RCHUNK * 448);

  auto zero = [&](float* p, size_t n) {
    zero_kernel<<<(unsigned)((n + 255) / 256), 256, 0, stream>>>(p, (int)n);
  };

  cg_init_kernel<<<16, 64, 0, stream>>>(cgw);
  geom_kernel<<<NEDGES / 256, 256, 0, stream>>>(xp, srcp, dstp, wp, Yb, efb);
  zero(hA, (size_t)NNODES * 256);
  embed_kernel<<<NNODES / 128, 256, 0, stream>>>(fx, in(EMBW), in(EMBB), hA);

  auto launch_conv = [&](int a, int b, float* outb, int e0) {
    dim3 g(RCHUNK / 256), t(256);
    const float* cgp = cgw + cg_pair_off(a, b);
    switch (a * 4 + b) {
#define CC(AA, BB) case AA * 4 + BB: \
      conv_pair_kernel<AA, BB><<<g, t, 0, stream>>>(Rbuf, Yb, hA, srcp, cgp, outb, e0, RCHUNK); break;
      CC(0, 0) CC(0, 1) CC(0, 2) CC(0, 3)
      CC(1, 0) CC(1, 1) CC(1, 2) CC(1, 3)
      CC(2, 0) CC(2, 1) CC(2, 2) CC(2, 3)
      CC(3, 0) CC(3, 1) CC(3, 2) CC(3, 3)
#undef CC
    }
  };
  auto run_pair = [&](RP r, int a, int b, float* outb) {
    radial_h2_kernel<<<NEDGES / 128, 256, 0, stream>>>(efb, r.W1, r.b1, r.W2, r.b2, H2b);
    int nf = 2 * (a < b ? a : b) + 1;
    int cols = 64 * nf;
    for (int ch = 0; ch < NEDGES / RCHUNK; ++ch) {
      int e0 = ch * RCHUNK;
      radial_r_kernel<<<RCHUNK / 128, 256, 0, stream>>>(H2b, r.W3, r.b3, Rbuf, cols, e0);
      launch_conv(a, b, outb, e0);
    }
  };

  for (int layer = 0; layer < 2; ++layer) {
    int ndin = layer ? 4 : 1;
    zero(vbuf, (size_t)NEDGES * 64);
    zero(kbuf, (size_t)NEDGES * 64);
    for (int a = 0; a < ndin; ++a)
      for (int b = 0; b < 4; ++b) {
        int pidx = layer ? (a * 4 + b) : b;
        run_pair(rp(VB[layer] + 6 * pidx), a, b, vbuf);
      }
    for (int a = 0; a < ndin; ++a)
      for (int b = 0; b < ndin; ++b) {
        int pidx = layer ? (a * 4 + b) : 0;
        run_pair(rp(KB[layer] + 6 * pidx), a, b, kbuf);
      }
    const float* q0 = in(QB[layer]);
    const float* q1 = ndin > 1 ? in(QB[layer] + 1) : q0;
    const float* q2 = ndin > 2 ? in(QB[layer] + 2) : q0;
    const float* q3 = ndin > 3 ? in(QB[layer] + 3) : q0;
    q_kernel<<<NNODES / 256, 256, 0, stream>>>(hA, q0, q1, q2, q3, qbuf, ndin);
    int mtot = ndin * ndin;
    float inv = 1.f / sqrtf(2.f * (float)mtot);
    logit_kernel<<<NEDGES / 256, 256, 0, stream>>>(kbuf, qbuf, dstp, lgb, mtot, inv);
    softmax_kernel<<<NNODES / 256, 256, 0, stream>>>(lgb);
    for (int d = 0; d < 4; ++d) {
      const float* pj = in(PROJB[layer] + d);
      const float* nw = in(NORMB[layer] + 2 * d);
      const float* nb = in(NORMB[layer] + 2 * d + 1);
      dim3 g(NNODES / 256), t(256);
      switch (d) {
        case 0: agg_kernel<0><<<g, t, 0, stream>>>(vbuf, lgb, hA, pj, nw, nb, hB, ndin); break;
        case 1: agg_kernel<1><<<g, t, 0, stream>>>(vbuf, lgb, hA, pj, nw, nb, hB, ndin); break;
        case 2: agg_kernel<2><<<g, t, 0, stream>>>(vbuf, lgb, hA, pj, nw, nb, hB, ndin); break;
        case 3: agg_kernel<3><<<g, t, 0, stream>>>(vbuf, lgb, hA, pj, nw, nb, hB, ndin); break;
      }
    }
    float* tswap = hA; hA = hB; hB = tswap;
  }

  // final pconv (dout=[0], co=64, nf=1) as outer-product WMMA GEMM
  zero(msgb, (size_t)NEDGES * 64);
  for (int a = 0; a < 4; ++a) {
    RP r = rp(finalPair(a));
    radial_h2_kernel<<<NEDGES / 128, 256, 0, stream>>>(efb, r.W1, r.b1, r.W2, r.b2, H2b);
    const float* cgp = cgw + cg_pair_off(a, 0);
    dim3 g(NEDGES / 256), t(256);
    switch (a) {
      case 0: tmp_kernel<0><<<g, t, 0, stream>>>(hA, srcp, Yb, cgp, tmpb); break;
      case 1: tmp_kernel<1><<<g, t, 0, stream>>>(hA, srcp, Yb, cgp, tmpb); break;
      case 2: tmp_kernel<2><<<g, t, 0, stream>>>(hA, srcp, Yb, cgp, tmpb); break;
      case 3: tmp_kernel<3><<<g, t, 0, stream>>>(hA, srcp, Yb, cgp, tmpb); break;
    }
    final_outer_kernel<<<NEDGES / 128, 256, 0, stream>>>(H2b, tmpb, r.W3, r.b3, msgb);
  }
  zero((float*)d_out, 64);
  final_node_kernel<<<NNODES / 256, 256, 0, stream>>>(msgb, hA, in(SELFW), (float*)d_out);
}
